// Model_39676907883783
// MI455X (gfx1250) — matrix operand order repaired and checked by reading, not compiled, not run on hardware
//
#include <hip/hip_runtime.h>
#include <hip/hip_bf16.h>
#include <stdint.h>

// ---------------------------------------------------------------------------
// Fused attention for (b=4, h=16, s=1024) fp32 tensors on gfx1250 (MI455X).
//   qk = (A@J + B) * (1/32) + mask ; attn = softmax(qk) ; out = attn @ P
// Per block = 64 rows of one (b,h) head, 4 waves of 32.
//   Phase 1: S = A@J via v_wmma_f32_16x16x32_bf16 (A tile cached bf16 in LDS,
//            J register-staged + double-buffered transposed bf16 in LDS so
//            global latency overlaps WMMA), online softmax, unnormalized
//            exp() rows stored bf16 in LDS with per-64-col-chunk running max.
//            Mask tile prefetched in the last chunk's (empty) load slot.
//   Rescale: fix up stale chunks with exp(m_chunk - m_final)/l in LDS.
//   Phase 2: out = attn @ P, attn frags straight from LDS, P pipelined same.
// f32->bf16 uses packed v_cvt_pk_bf16_f32 (1 VALU op per 2 elements).
// LDS ~297KB of the 320KB/WGP (dynamic shared memory).
// ---------------------------------------------------------------------------

typedef __attribute__((ext_vector_type(16))) __bf16 v16bf;
typedef __attribute__((ext_vector_type(8)))  float  v8f;

#define S_DIM    1024
#define BM       64          // rows per block
#define BN       64          // column chunk (softmax chunk / output tile)
#define KC       128         // k-depth staged per LDS refill
#define NTHREADS 128         // 4 waves of 32
#define LDA      (S_DIM + 8) // padded row stride (bf16 elems) for sA / sAttn
#define LDJ      (KC + 8)    // padded row stride for transposed J/P staging
#define NLD      ((KC * BN) / (4 * NTHREADS))   // float4 staged per thread = 16

// LDS partition (bytes)
#define OFF_A    0
#define OFF_ATTN (OFF_A    + BM * LDA * 2)        // 132096
#define OFF_JT   (OFF_ATTN + BM * LDA * 2)        // 264192 (double buffered)
#define OFF_M    (OFF_JT   + 2 * BN * LDJ * 2)    // 299008
#define OFF_ROW  (OFF_M    + BM * 16 * 4)         // 303104
#define SMEM_TOT (OFF_ROW  + BM * 2 * 4)          // 303616

union FragBF {
    v16bf v;
    uint4 q[2];
};

// packed f32x2 -> bf16x2 (round-to-nearest-even) in one VALU op
__device__ __forceinline__ unsigned cvt_pk_bf16(float lo, float hi) {
    unsigned r;
    asm("v_cvt_pk_bf16_f32 %0, %1, %2" : "=v"(r) : "v"(lo), "v"(hi));
    return r;
}

__device__ __forceinline__ unsigned short f2bf(float f) {
    union { float f; unsigned u; } x;
    x.f = f;
    unsigned r = x.u + 0x7fffu + ((x.u >> 16) & 1u);  // round-to-nearest-even
    return (unsigned short)(r >> 16);
}

__device__ __forceinline__ float bits2f(unsigned u) {
    float f;
    __builtin_memcpy(&f, &u, 4);
    return f;
}

extern "C" __global__ void __launch_bounds__(NTHREADS, 1)
fused_attn_kernel(const float* __restrict__ A, const float* __restrict__ J,
                  const float* __restrict__ Bm, const float* __restrict__ P,
                  const float* __restrict__ mask, float* __restrict__ out)
{
    extern __shared__ char smem[];
    unsigned short* sA    = (unsigned short*)(smem + OFF_A);
    unsigned short* sAttn = (unsigned short*)(smem + OFF_ATTN);
    unsigned short* sJt   = (unsigned short*)(smem + OFF_JT);
    float*          sM    = (float*)(smem + OFF_M);
    float*          sRow  = (float*)(smem + OFF_ROW);

    const int tid   = threadIdx.x;
    const int lane  = tid & 31;
    const int wave  = tid >> 5;
    const int khalf = lane >> 4;   // 0: lanes 0-15, 1: lanes 16-31
    const int ln16  = lane & 15;

    const int bh   = blockIdx.y;
    const int row0 = blockIdx.x * BM;
    const size_t bhOff = (size_t)bh * S_DIM * S_DIM;
    const float scale = 0.03125f;  // 1/sqrt(1024)
    const int rowL0 = wave * 16;   // this wave's local row base

    // per-thread staging coordinates (fixed across chunks)
    int stK[NLD], stN[NLD];
#pragma unroll
    for (int it = 0; it < NLD; ++it) {
        const int f4 = it * NTHREADS + tid;
        stK[it] = f4 >> 4;          // 0..KC-1  (16 float4 per 64-wide row)
        stN[it] = (f4 & 15) * 4;    // 0..60
    }

    // ---- Stage A tile (BM x S) fp32 -> bf16 into sA --------------------
    for (int idx = tid; idx < BM * (S_DIM / 4); idx += NTHREADS) {
        int r  = idx >> 8;                 // S_DIM/4 = 256 per row
        int c4 = (idx & 255) * 4;
        const float4 f = *(const float4*)(A + bhOff + (size_t)(row0 + r) * S_DIM + c4);
        uint2 u;
        u.x = cvt_pk_bf16(f.x, f.y);
        u.y = cvt_pk_bf16(f.z, f.w);
        *(uint2*)(sA + r * LDA + c4) = u;
    }
    __syncthreads();

    // ---- Phase 1: S = (A@J + B)*scale + mask, online softmax -----------
    float m_run[8], l_run[8];
#pragma unroll
    for (int r = 0; r < 8; ++r) { m_run[r] = -3.0e38f; l_run[r] = 0.0f; }

    for (int jb = 0; jb < S_DIM; jb += BN) {
        // prefetch bias tile for this jb; consumed ~2000 cycles later
        float biasv[4][8];
#pragma unroll
        for (int t = 0; t < 4; ++t) {
            const int col = jb + t * 16 + ln16;
#pragma unroll
            for (int r = 0; r < 8; ++r) {
                const int rg = row0 + rowL0 + r + 8 * khalf;
                biasv[t][r] = Bm[bhOff + (size_t)rg * S_DIM + col];
            }
        }
        float maskv[4][8];   // filled during the last k-chunk's load slot

        v8f acc[4];
#pragma unroll
        for (int t = 0; t < 4; ++t)
#pragma unroll
            for (int e = 0; e < 8; ++e) acc[t][e] = 0.0f;

        // pipeline prologue: global-load chunk 0 into registers
        float4 ld[NLD];
#pragma unroll
        for (int it = 0; it < NLD; ++it)
            ld[it] = *(const float4*)(J + bhOff + (size_t)stK[it] * S_DIM + jb + stN[it]);

        int buf = 0;
        for (int kc = 0; kc < S_DIM; kc += KC) {
            // convert + store staged chunk into sJt[buf] (transposed [n][k])
            unsigned short* sj = sJt + buf * (BN * LDJ);
#pragma unroll
            for (int it = 0; it < NLD; ++it) {
                const unsigned u01 = cvt_pk_bf16(ld[it].x, ld[it].y);
                const unsigned u23 = cvt_pk_bf16(ld[it].z, ld[it].w);
                sj[(stN[it] + 0) * LDJ + stK[it]] = (unsigned short)(u01 & 0xffffu);
                sj[(stN[it] + 1) * LDJ + stK[it]] = (unsigned short)(u01 >> 16);
                sj[(stN[it] + 2) * LDJ + stK[it]] = (unsigned short)(u23 & 0xffffu);
                sj[(stN[it] + 3) * LDJ + stK[it]] = (unsigned short)(u23 >> 16);
            }
            __syncthreads();

            // issue next chunk's global loads (or, on the last chunk, the
            // mask tile); either way they complete under the WMMAs below
            if (kc + KC < S_DIM) {
#pragma unroll
                for (int it = 0; it < NLD; ++it)
                    ld[it] = *(const float4*)(J + bhOff +
                             (size_t)(kc + KC + stK[it]) * S_DIM + jb + stN[it]);
            } else {
#pragma unroll
                for (int t = 0; t < 4; ++t) {
                    const int col = jb + t * 16 + ln16;
#pragma unroll
                    for (int r = 0; r < 8; ++r) {
                        const int rg = row0 + rowL0 + r + 8 * khalf;
                        maskv[t][r] = mask[(size_t)rg * S_DIM + col];
                    }
                }
            }

#pragma unroll
            for (int kk = 0; kk < KC; kk += 32) {
                FragBF a;
                const unsigned short* ab = sA + (rowL0 + ln16) * LDA + kc + kk + 8 * khalf;
                a.q[0] = *(const uint4*)(ab);
                a.q[1] = *(const uint4*)(ab + 16);
                FragBF b[4];
#pragma unroll
                for (int t = 0; t < 4; ++t) {
                    const unsigned short* bb = sj + (t * 16 + ln16) * LDJ + kk + 8 * khalf;
                    b[t].q[0] = *(const uint4*)(bb);
                    b[t].q[1] = *(const uint4*)(bb + 16);
                }
#pragma unroll
                for (int t = 0; t < 4; ++t)
                    acc[t] = __builtin_amdgcn_wmma_f32_16x16x32_bf16(
                        false, a.v, false, b[t].v, (short)0, acc[t], false, false);
            }
            buf ^= 1;
        }

        // epilogue: bias + mask + scale, chunk max, exp, store to sAttn
        float pv[4][8];
        float chmax[8];
#pragma unroll
        for (int r = 0; r < 8; ++r) chmax[r] = -3.0e38f;
#pragma unroll
        for (int t = 0; t < 4; ++t) {
#pragma unroll
            for (int r = 0; r < 8; ++r) {
                float v = (acc[t][r] + biasv[t][r]) * scale + maskv[t][r];
                pv[t][r] = v;
                chmax[r] = fmaxf(chmax[r], v);
            }
        }
#pragma unroll
        for (int r = 0; r < 8; ++r) {
            float m = chmax[r];
            m = fmaxf(m, __shfl_xor(m, 1, 32));
            m = fmaxf(m, __shfl_xor(m, 2, 32));
            m = fmaxf(m, __shfl_xor(m, 4, 32));
            m = fmaxf(m, __shfl_xor(m, 8, 32));   // reduce within 16-lane half
            float mnew = fmaxf(m_run[r], m);
            float corr = __expf(m_run[r] - mnew);
            float psum = 0.0f;
#pragma unroll
            for (int t = 0; t < 4; ++t) {
                float p = __expf(pv[t][r] - mnew);
                psum += p;
                sAttn[(rowL0 + r + 8 * khalf) * LDA + jb + t * 16 + ln16] = f2bf(p);
            }
            psum += __shfl_xor(psum, 1, 32);
            psum += __shfl_xor(psum, 2, 32);
            psum += __shfl_xor(psum, 4, 32);
            psum += __shfl_xor(psum, 8, 32);
            l_run[r] = l_run[r] * corr + psum;
            m_run[r] = mnew;
            if (ln16 == 0)
                sM[(rowL0 + r + 8 * khalf) * 16 + (jb >> 6)] = mnew;
        }
    }

#pragma unroll
    for (int r = 0; r < 8; ++r) {
        if (ln16 == 0) {
            const int rl = rowL0 + r + 8 * khalf;
            sRow[rl * 2 + 0] = m_run[r];
            sRow[rl * 2 + 1] = 1.0f / l_run[r];
        }
    }
    __syncthreads();

    // ---- Rescale stale chunks: attn *= exp(m_chunk - m_final) / l ------
    {
        const int rl = tid >> 1;                 // 2 threads per row
        const int c0 = (tid & 1) * 512;
        const float mfin = sRow[rl * 2 + 0];
        const float linv = sRow[rl * 2 + 1];
        unsigned short* rowp = sAttn + rl * LDA;
        for (int ch = 0; ch < 8; ++ch) {
            const float f = __expf(sM[rl * 16 + ((c0 + ch * 64) >> 6)] - mfin) * linv;
            uint4* p4 = (uint4*)(rowp + c0 + ch * 64);
#pragma unroll 2
            for (int q = 0; q < 8; ++q) {
                uint4 u = p4[q];
                unsigned* uw = (unsigned*)&u;
#pragma unroll
                for (int e = 0; e < 4; ++e) {
                    const float lo = bits2f(uw[e] << 16) * f;
                    const float hi = bits2f(uw[e] & 0xffff0000u) * f;
                    uw[e] = cvt_pk_bf16(lo, hi);
                }
                p4[q] = u;
            }
        }
    }
    __syncthreads();

    // ---- Phase 2: out = attn @ P (same double-buffered pipeline) -------
    for (int nb = 0; nb < S_DIM; nb += BN) {
        v8f acc[4];
#pragma unroll
        for (int t = 0; t < 4; ++t)
#pragma unroll
            for (int e = 0; e < 8; ++e) acc[t][e] = 0.0f;

        float4 ld[NLD];
#pragma unroll
        for (int it = 0; it < NLD; ++it)
            ld[it] = *(const float4*)(P + bhOff + (size_t)stK[it] * S_DIM + nb + stN[it]);

        int buf = 0;
        for (int kc = 0; kc < S_DIM; kc += KC) {
            unsigned short* sj = sJt + buf * (BN * LDJ);
#pragma unroll
            for (int it = 0; it < NLD; ++it) {
                const unsigned u01 = cvt_pk_bf16(ld[it].x, ld[it].y);
                const unsigned u23 = cvt_pk_bf16(ld[it].z, ld[it].w);
                sj[(stN[it] + 0) * LDJ + stK[it]] = (unsigned short)(u01 & 0xffffu);
                sj[(stN[it] + 1) * LDJ + stK[it]] = (unsigned short)(u01 >> 16);
                sj[(stN[it] + 2) * LDJ + stK[it]] = (unsigned short)(u23 & 0xffffu);
                sj[(stN[it] + 3) * LDJ + stK[it]] = (unsigned short)(u23 >> 16);
            }
            __syncthreads();

            if (kc + KC < S_DIM) {
#pragma unroll
                for (int it = 0; it < NLD; ++it)
                    ld[it] = *(const float4*)(P + bhOff +
                             (size_t)(kc + KC + stK[it]) * S_DIM + nb + stN[it]);
            }

#pragma unroll
            for (int kk = 0; kk < KC; kk += 32) {
                FragBF a;
                const unsigned short* ab = sAttn + (rowL0 + ln16) * LDA + kc + kk + 8 * khalf;
                a.q[0] = *(const uint4*)(ab);
                a.q[1] = *(const uint4*)(ab + 16);
                FragBF b[4];
#pragma unroll
                for (int t = 0; t < 4; ++t) {
                    const unsigned short* bb = sj + (t * 16 + ln16) * LDJ + kk + 8 * khalf;
                    b[t].q[0] = *(const uint4*)(bb);
                    b[t].q[1] = *(const uint4*)(bb + 16);
                }
#pragma unroll
                for (int t = 0; t < 4; ++t)
                    acc[t] = __builtin_amdgcn_wmma_f32_16x16x32_bf16(
                        false, a.v, false, b[t].v, (short)0, acc[t], false, false);
            }
            buf ^= 1;
        }

#pragma unroll
        for (int t = 0; t < 4; ++t) {
            const int col = nb + t * 16 + ln16;
#pragma unroll
            for (int r = 0; r < 8; ++r) {
                const int rg = row0 + rowL0 + r + 8 * khalf;
                out[bhOff + (size_t)rg * S_DIM + col] = acc[t][r];
            }
        }
    }
}

extern "C" void kernel_launch(void* const* d_in, const int* in_sizes, int n_in,
                              void* d_out, int out_size, void* d_ws, size_t ws_size,
                              hipStream_t stream) {
    (void)in_sizes; (void)n_in; (void)d_ws; (void)ws_size; (void)out_size;
    const float* A    = (const float*)d_in[0];
    const float* J    = (const float*)d_in[1];
    const float* B    = (const float*)d_in[2];
    const float* P    = (const float*)d_in[3];
    const float* mask = (const float*)d_in[4];
    float* out = (float*)d_out;

    // Allow the ~297KB dynamic-LDS workgroup (CDNA5 WGP has 320KB).
    hipFuncSetAttribute((const void*)fused_attn_kernel,
                        hipFuncAttributeMaxDynamicSharedMemorySize,
                        (int)SMEM_TOT);

    dim3 grid(S_DIM / BM, 64);   // 16 row-blocks x (b*h)=64 heads
    dim3 block(NTHREADS);
    fused_attn_kernel<<<grid, block, SMEM_TOT, stream>>>(A, J, B, P, mask, out);
}
